// GAP_net_53772990546012
// MI455X (gfx1250) — matrix operand order repaired and checked by reading, not compiled, not run on hardware
//
#include <hip/hip_runtime.h>
#include <hip/hip_bf16.h>
#include <math.h>

typedef __attribute__((ext_vector_type(16))) _Float16 v16h;
typedef __attribute__((ext_vector_type(8)))  _Float16 v8h;
typedef __attribute__((ext_vector_type(8)))  float    v8f;

#define EW_BLOCK 256

// ---------------------------------------------------------------------------
// Implicit-GEMM conv (SAME, zero pad) via V_WMMA_F32_16X16X32_F16.
// Activations: f16 NHWC, channels padded to multiple of 32 (pad channels = 0).
// Weights: pre-packed f16 in exact B-fragment order (one v16h load per lane).
// One wave computes 32 pixels x 16 out-channels (M-blocked x2): the B fragment
// is loaded once per K-chunk and feeds two independent WMMA accumulator chains
// (better issue-traffic/FLOP and XDL latency hiding).
// A frag (16x32): lanes<16 halves0..7->K0..7, 8..15->K16..23; lanes>=16 +8.
//   => per chunk: two contiguous v8h loads at ci = cb+hi*8 and cb+16+hi*8.
// B frag (32x16): lane n%16 = out ch; element e -> K = (e&7) + (e>>3)*16 + hi*8 (packed).
// D (16x16 f32): N = lane%16, row M = r + hi*8.
// ---------------------------------------------------------------------------
template <int KS>
__global__ __launch_bounds__(32) void conv_wmma_nhwc(
    const _Float16* __restrict__ in,   // [B,H,W,Cip]
    const _Float16* __restrict__ wB,   // packed fragments
    const float* __restrict__ bias,    // [Cout]
    _Float16* __restrict__ out,        // [B,H,W,Cop]
    int Cip, int nch, int Cout, int Cop, int H, int W, int relu)
{
  constexpr int KK  = KS * KS;
  constexpr int PAD = KS / 2;

  const int t      = threadIdx.x;
  const int lane16 = t & 15;
  const int hi     = t >> 4;
  const int HW     = H * W;

  const int mt = blockIdx.x;
  const int nt = blockIdx.y;
  const int b  = blockIdx.z;

  // two 16-pixel subtiles per wave
  const int p0  = mt * 32 + lane16;
  const int p1  = p0 + 16;
  const int ph0 = p0 / W;                 // loop-invariant divisions
  const int pw0 = p0 - ph0 * W;
  const int ph1 = p1 / W;
  const int pw1 = p1 - ph1 * W;
  const bool pv0 = (p0 < HW);
  const bool pv1 = (p1 < HW);
  const int n = nt * 16 + lane16;         // out channel (B/D column)

  const _Float16* __restrict__ inb = in + (size_t)b * HW * Cip;
  const _Float16* __restrict__ wnt = wB + ((size_t)nt * KK * nch) * 512 + t * 16;

  v16h az;
#pragma unroll
  for (int e = 0; e < 16; ++e) az[e] = (_Float16)0.f;

  v8f acc0 = {0.f, 0.f, 0.f, 0.f, 0.f, 0.f, 0.f, 0.f};
  v8f acc1 = {0.f, 0.f, 0.f, 0.f, 0.f, 0.f, 0.f, 0.f};

  for (int tap = 0; tap < KK; ++tap) {
    const int kh = tap / KS;              // constexpr divisor
    const int kw = tap - kh * KS;
    const int ih0 = ph0 + kh - PAD, iw0 = pw0 + kw - PAD;
    const int ih1 = ph1 + kh - PAD, iw1 = pw1 + kw - PAD;
    const bool sv0 = pv0 && ((unsigned)ih0 < (unsigned)H) && ((unsigned)iw0 < (unsigned)W);
    const bool sv1 = pv1 && ((unsigned)ih1 < (unsigned)H) && ((unsigned)iw1 < (unsigned)W);
    const _Float16* __restrict__ ap0 =
        inb + ((size_t)(sv0 ? ih0 : 0) * W + (sv0 ? iw0 : 0)) * Cip + hi * 8;
    const _Float16* __restrict__ ap1 =
        inb + ((size_t)(sv1 ? ih1 : 0) * W + (sv1 ? iw1 : 0)) * Cip + hi * 8;
    const _Float16* __restrict__ wtap = wnt + (size_t)tap * nch * 512;
    for (int cbi = 0; cbi < nch; ++cbi) {
      v16h afrag0, afrag1;
      if (sv0) {
        v8h lo = *(const v8h*)(ap0 + cbi * 32);
        v8h hg = *(const v8h*)(ap0 + cbi * 32 + 16);
#pragma unroll
        for (int e = 0; e < 8; ++e) { afrag0[e] = lo[e]; afrag0[e + 8] = hg[e]; }
      } else {
        afrag0 = az;
      }
      if (sv1) {
        v8h lo = *(const v8h*)(ap1 + cbi * 32);
        v8h hg = *(const v8h*)(ap1 + cbi * 32 + 16);
#pragma unroll
        for (int e = 0; e < 8; ++e) { afrag1[e] = lo[e]; afrag1[e + 8] = hg[e]; }
      } else {
        afrag1 = az;
      }
      v16h bfrag = *(const v16h*)(wtap + (size_t)cbi * 512);  // shared by both
      acc0 = __builtin_amdgcn_wmma_f32_16x16x32_f16(
          false, afrag0, false, bfrag, (short)0, acc0, false, false);
      acc1 = __builtin_amdgcn_wmma_f32_16x16x32_f16(
          false, afrag1, false, bfrag, (short)0, acc1, false, false);
    }
  }

  if (n < Cout) {
    const float bv = bias[n];
    _Float16* __restrict__ outb = out + (size_t)b * HW * Cop;
#pragma unroll
    for (int r = 0; r < 8; ++r) {
      const int m0 = mt * 32 + r + hi * 8;
      const int m1 = m0 + 16;
      if (m0 < HW) {
        float v = acc0[r] + bv;
        if (relu) v = fmaxf(v, 0.f);
        outb[(size_t)m0 * Cop + n] = (_Float16)v;
      }
      if (m1 < HW) {
        float v = acc1[r] + bv;
        if (relu) v = fmaxf(v, 0.f);
        outb[(size_t)m1 * Cop + n] = (_Float16)v;
      }
    }
  }
}

// Pack weights [Cout,Cin,KS,KS] (f32) -> f16 fragments, lane-major:
// wB[(((nt*KK + tap)*nch + cbi)*32 + t)*16 + e] = w[n][ci][tap],
//   n = nt*16 + (t&15), ci = cbi*32 + (e&7) + (e>>3)*16 + (t>>4)*8, zero if OOB.
__global__ __launch_bounds__(EW_BLOCK) void repack_w_kernel(
    const float* __restrict__ w, _Float16* __restrict__ wB,
    int Cout, int Cin, int KK, int nch, long n)
{
  long i = (long)blockIdx.x * EW_BLOCK + threadIdx.x;
  if (i >= n) return;
  int e = (int)(i & 15);
  int t = (int)((i >> 4) & 31);
  long chunk = i >> 9;
  int cbi = (int)(chunk % nch);
  long c2  = chunk / nch;
  int tap  = (int)(c2 % KK);
  int nt   = (int)(c2 / KK);
  int nn = nt * 16 + (t & 15);
  int ci = cbi * 32 + (e & 7) + (e >> 3) * 16 + (t >> 4) * 8;
  float v = (nn < Cout && ci < Cin) ? w[((size_t)nn * Cin + ci) * KK + tap] : 0.f;
  wB[i] = (_Float16)v;
}

// ---------------------------------------------------------------------------
// GAP / shift / pooling / concat / upsample kernels
// ---------------------------------------------------------------------------
__global__ __launch_bounds__(EW_BLOCK) void at_init_kernel(
    const float* __restrict__ y, const float* __restrict__ Phi,
    float* __restrict__ x, int C, int HW, long n)
{
  long i = (long)blockIdx.x * EW_BLOCK + threadIdx.x;
  if (i >= n) return;
  int p = (int)(i % HW);
  int c = (int)((i / HW) % C);
  int b = (int)(i / ((long)HW * C));
  x[i] = y[(size_t)b * HW + p] * Phi[(size_t)c * HW + p];
}

__global__ __launch_bounds__(EW_BLOCK) void gap_step_kernel(
    float* __restrict__ x, const float* __restrict__ y,
    const float* __restrict__ Phi, const float* __restrict__ Phi_s,
    int C, int HW, long n)
{
  long i = (long)blockIdx.x * EW_BLOCK + threadIdx.x;
  if (i >= n) return;
  int p = (int)(i % HW);
  int b = (int)(i / HW);
  float yb = 0.f;
  for (int c = 0; c < C; ++c)
    yb += x[((size_t)b * C + c) * HW + p] * Phi[(size_t)c * HW + p];
  float s = (y[i] - yb) / Phi_s[p];
  for (int c = 0; c < C; ++c)
    x[((size_t)b * C + c) * HW + p] += s * Phi[(size_t)c * HW + p];
}

// shift_back(-1) + reflect pad 310->312, f32 NCHW -> f16 NHWC (Cpad, zero-fill)
__global__ __launch_bounds__(EW_BLOCK) void shift_pad_f16_kernel(
    const float* __restrict__ x, _Float16* __restrict__ xin,
    int C, int Cpad, int H, int W, int WP, long n)
{
  long i = (long)blockIdx.x * EW_BLOCK + threadIdx.x;
  if (i >= n) return;
  int c = (int)(i % Cpad);
  long q = i / Cpad;
  int w = (int)(q % WP); q /= WP;
  int h = (int)(q % H);
  int b = (int)(q / H);
  float v = 0.f;
  if (c < C) {
    int j = (w < W) ? w : ((w == W) ? (W - 2) : (W - 3));  // reflect pad
    int src = j + 2 * c;                                   // shift_back: j + STEP*c
    if (src >= W) src -= W;
    v = x[(((size_t)b * C + c) * H + h) * W + src];
  }
  xin[i] = (_Float16)v;
}

// 2x2 max pool, f16 NHWC
__global__ __launch_bounds__(EW_BLOCK) void maxpool2_f16_kernel(
    const _Float16* __restrict__ src, _Float16* __restrict__ dst,
    int C, int H, int W, long n)
{
  long i = (long)blockIdx.x * EW_BLOCK + threadIdx.x;
  if (i >= n) return;
  int Ho = H >> 1, Wo = W >> 1;
  int c = (int)(i % C);
  long q = i / C;
  int wo = (int)(q % Wo); q /= Wo;
  int ho = (int)(q % Ho);
  int b  = (int)(q / Ho);
  const _Float16* s = src + (((size_t)b * H + 2 * ho) * W + 2 * wo) * C + c;
  float a0 = (float)s[0], a1 = (float)s[C];
  float a2 = (float)s[(size_t)W * C], a3 = (float)s[(size_t)W * C + C];
  dst[i] = (_Float16)fmaxf(fmaxf(a0, a1), fmaxf(a2, a3));
}

// ConvTranspose2d(kernel=2, stride=2) + ReLU, f16 NHWC. wgt f32 [Cin,Cout,2,2]
__global__ __launch_bounds__(EW_BLOCK) void upconv2_f16_kernel(
    const _Float16* __restrict__ in, const float* __restrict__ wgt,
    const float* __restrict__ bias, _Float16* __restrict__ out,
    int Cin, int Cout, int h, int w, long n)
{
  long i = (long)blockIdx.x * EW_BLOCK + threadIdx.x;
  if (i >= n) return;
  int W2 = 2 * w, H2 = 2 * h;
  int o  = (int)(i % Cout);
  long q = i / Cout;
  int x  = (int)(q % W2); q /= W2;
  int yy = (int)(q % H2);
  int b  = (int)(q / H2);
  int iy = yy >> 1, ix = x >> 1, dy = yy & 1, dx = x & 1;
  const _Float16* inp = in + (((size_t)b * h + iy) * w + ix) * Cin;
  float s = bias[o];
  for (int c = 0; c < Cin; ++c)
    s += (float)inp[c] * wgt[(((size_t)c * Cout + o) * 2 + dy) * 2 + dx];
  out[i] = (_Float16)fmaxf(s, 0.f);
}

// concat along channels, f16 NHWC: dst[..., coff + c] = src[..., c]
__global__ __launch_bounds__(EW_BLOCK) void copych_f16_kernel(
    const _Float16* __restrict__ src, _Float16* __restrict__ dst,
    int Cs, int Cd, int coff, long HW, long n)
{
  long i = (long)blockIdx.x * EW_BLOCK + threadIdx.x;
  if (i >= n) return;
  int c = (int)(i % Cs);
  long q = i / Cs;
  long p = q % HW;
  long b = q / HW;
  dst[((size_t)b * HW + p) * Cd + coff + c] = src[i];
}

// shift(+1)(tanh(last)) + x, in place:  x[(j+2c)%W] += tanh(last[j])
// (uses shift(+1)(unet_out + shift_back(x)) == shift(+1)(unet_out) + x,
//  so the f32 GAP state never passes through f16.)
__global__ __launch_bounds__(EW_BLOCK) void final_add_shift_kernel(
    const _Float16* __restrict__ last,  // [B,H,WP,Cop]
    float* __restrict__ x,              // [B,C,H,W]
    int C, int Cop, int H, int W, int WP, long n)
{
  long i = (long)blockIdx.x * EW_BLOCK + threadIdx.x;
  if (i >= n) return;
  int j = (int)(i % W);
  long q = i / W;
  int h = (int)(q % H); q /= H;
  int c = (int)(q % C);
  int b = (int)(q / C);
  float v = tanhf((float)last[(((size_t)b * H + h) * WP + j) * Cop + c]);
  int jj = j + 2 * c;
  if (jj >= W) jj -= W;
  x[(((size_t)b * C + c) * H + h) * W + jj] += v;
}

// ---------------------------------------------------------------------------
// Host orchestration
// ---------------------------------------------------------------------------
static inline int nblk(long n) { return (int)((n + EW_BLOCK - 1) / EW_BLOCK); }

static void launch_conv3(const _Float16* in, const _Float16* wB, const float* bias,
                         _Float16* out, int B, int Cip, int nch, int Cout, int Cop,
                         int H, int W, int relu, hipStream_t s)
{
  dim3 grid((H * W + 31) / 32, (Cout + 15) / 16, B);
  conv_wmma_nhwc<3><<<grid, dim3(32), 0, s>>>(in, wB, bias, out, Cip, nch, Cout, Cop, H, W, relu);
}

static void launch_conv1(const _Float16* in, const _Float16* wB, const float* bias,
                         _Float16* out, int B, int Cip, int nch, int Cout, int Cop,
                         int H, int W, int relu, hipStream_t s)
{
  dim3 grid((H * W + 31) / 32, (Cout + 15) / 16, B);
  conv_wmma_nhwc<1><<<grid, dim3(32), 0, s>>>(in, wB, bias, out, Cip, nch, Cout, Cop, H, W, relu);
}

extern "C" void kernel_launch(void* const* d_in, const int* in_sizes, int n_in,
                              void* d_out, int out_size, void* d_ws, size_t ws_size,
                              hipStream_t stream)
{
  const float* y     = (const float*)d_in[0];   // [2,256,310]
  const float* Phi   = (const float*)d_in[1];   // [1,28,256,310]
  const float* Phi_s = (const float*)d_in[2];   // [1,256,310]

  const int B = 2, C = 28, H = 256, W = 310, WP = 312;
  const int HW = H * W;

  // Param ordering detection (insertion vs JAX sorted-key flatten).
  bool insertion = (n_in > 3 && in_sizes[3] == 8064);
  static const int ins_idx[26] = {0,1,2,3,4,5,6,7,8,9,10,11,12,13,14,15,16,17,18,19,20,21,22,23,24,25};
  static const int srt_idx[26] = {2,0,3,1, 6,4,7,5, 10,8,11,9, 18,16,19,17, 14,12,15,13, 25,24,23,22,21,20};
  const int* idx = insertion ? ins_idx : srt_idx;

  const float* P[9][26];
  for (int k = 0; k < 9; ++k)
    for (int f = 0; f < 26; ++f)
      P[k][f] = (const float*)d_in[3 + 26 * k + idx[f]];

  enum { D1W1, D1B1, D1W2, D1B2, D2W1, D2B1, D2W2, D2B2, D3W1, D3B1, D3W2, D3B2,
         DU2W1, DU2B1, DU2W2, DU2B2, DU1W1, DU1B1, DU1W2, DU1B2,
         U2W, U2B, U1W, U1B, LW, LB };

  // Workspace arenas. f32 state first, then f16 NHWC activations + packed w.
  float* xb = (float*)d_ws;                               // [2,28,256,310] f32
  _Float16* hb = (_Float16*)(xb + (size_t)B * C * HW);    // 256B-aligned offset
  size_t ho = 0;
  _Float16* xin = hb + ho; ho += (size_t)B * H * WP * 32;   // [2,256,312,32]
  _Float16* c1  = hb + ho; ho += (size_t)B * H * WP * 32;   // [2,256,312,32]
  _Float16* c2  = hb + ho; ho += (size_t)B * 128 * 156 * 64;// [2,128,156,64]
  _Float16* Ab  = hb + ho; ho += (size_t)B * H * WP * 64;   // max: cat1 [2,256,312,64]
  _Float16* Bb  = hb + ho; ho += (size_t)B * H * WP * 32;   // [2,256,312,32]
  _Float16* wpk = hb + ho;                                  // packed weights (~3.9M halves)
  (void)ws_size;

  // Per-layer conv specs: {w idx, b idx, Cout, Cin, KK}
  struct LSpec { int widx, bidx, Cout, Cin, KK; };
  static const LSpec LS[11] = {
    {D1W1,D1B1, 32, 28,9}, {D1W2,D1B2, 32, 32,9},
    {D2W1,D2B1, 64, 32,9}, {D2W2,D2B2, 64, 64,9},
    {D3W1,D3B1,128, 64,9}, {D3W2,D3B2,128,128,9},
    {DU2W1,DU2B1,64,128,9}, {DU2W2,DU2B2,64, 64,9},
    {DU1W1,DU1B1,32, 64,9}, {DU1W2,DU1B2,32, 32,9},
    {LW,LB, 28, 32,1}};

  _Float16* wp[9][11];
  {
    size_t wo = 0;
    for (int k = 0; k < 9; ++k)
      for (int l = 0; l < 11; ++l) {
        int ntl = (LS[l].Cout + 15) / 16;
        int nch = (LS[l].Cin + 31) / 32;
        long sz = (long)ntl * LS[l].KK * nch * 512;
        wp[k][l] = wpk + wo; wo += (size_t)sz;
        repack_w_kernel<<<nblk(sz), EW_BLOCK, 0, stream>>>(
            P[k][LS[l].widx], wp[k][l], LS[l].Cout, LS[l].Cin, LS[l].KK, nch, sz);
      }
  }

  const long n_x = (long)B * C * HW;

  // x = At(y, Phi)
  at_init_kernel<<<nblk(n_x), EW_BLOCK, 0, stream>>>(y, Phi, xb, C, HW, n_x);

  for (int k = 0; k < 9; ++k) {
    long n_bhw = (long)B * HW;
    gap_step_kernel<<<nblk(n_bhw), EW_BLOCK, 0, stream>>>(xb, y, Phi, Phi_s, C, HW, n_bhw);

    // shift_back + reflect pad -> xin f16 NHWC [2,256,312,32] (ch 28..31 = 0)
    { long n = (long)B * H * WP * 32;
      shift_pad_f16_kernel<<<nblk(n), EW_BLOCK, 0, stream>>>(xb, xin, C, 32, H, W, WP, n); }

    // ---- encoder ----
    launch_conv3(xin, wp[k][0], P[k][D1B1], Ab, B, 32, 1, 32, 32, 256, 312, 1, stream);
    launch_conv3(Ab,  wp[k][1], P[k][D1B2], c1, B, 32, 1, 32, 32, 256, 312, 1, stream);
    { long n = (long)B * 128 * 156 * 32;
      maxpool2_f16_kernel<<<nblk(n), EW_BLOCK, 0, stream>>>(c1, Ab, 32, 256, 312, n); }
    launch_conv3(Ab,  wp[k][2], P[k][D2B1], Bb, B, 32, 1, 64, 64, 128, 156, 1, stream);
    launch_conv3(Bb,  wp[k][3], P[k][D2B2], c2, B, 64, 2, 64, 64, 128, 156, 1, stream);
    { long n = (long)B * 64 * 78 * 64;
      maxpool2_f16_kernel<<<nblk(n), EW_BLOCK, 0, stream>>>(c2, Ab, 64, 128, 156, n); }
    launch_conv3(Ab,  wp[k][4], P[k][D3B1], Bb, B, 64, 2, 128, 128, 64, 78, 1, stream);
    launch_conv3(Bb,  wp[k][5], P[k][D3B2], Ab, B, 128, 4, 128, 128, 64, 78, 1, stream);

    // ---- decoder level 2 ----
    { long n = (long)B * 128 * 156 * 64;
      upconv2_f16_kernel<<<nblk(n), EW_BLOCK, 0, stream>>>(Ab, P[k][U2W], P[k][U2B], Bb,
                                                           128, 64, 64, 78, n); }
    { long n = (long)B * 128 * 156 * 64; long hw = 128 * 156;
      copych_f16_kernel<<<nblk(n), EW_BLOCK, 0, stream>>>(Bb, Ab, 64, 128, 0,  hw, n);
      copych_f16_kernel<<<nblk(n), EW_BLOCK, 0, stream>>>(c2, Ab, 64, 128, 64, hw, n); }
    launch_conv3(Ab, wp[k][6], P[k][DU2B1], Bb, B, 128, 4, 64, 64, 128, 156, 1, stream);
    launch_conv3(Bb, wp[k][7], P[k][DU2B2], Ab, B, 64, 2, 64, 64, 128, 156, 1, stream);

    // ---- decoder level 1 ----
    { long n = (long)B * 256 * 312 * 32;
      upconv2_f16_kernel<<<nblk(n), EW_BLOCK, 0, stream>>>(Ab, P[k][U1W], P[k][U1B], Bb,
                                                           64, 32, 128, 156, n); }
    { long n = (long)B * 256 * 312 * 32; long hw = 256 * 312;
      copych_f16_kernel<<<nblk(n), EW_BLOCK, 0, stream>>>(Bb, Ab, 32, 64, 0,  hw, n);
      copych_f16_kernel<<<nblk(n), EW_BLOCK, 0, stream>>>(c1, Ab, 32, 64, 32, hw, n); }
    launch_conv3(Ab, wp[k][8], P[k][DU1B1], Bb, B, 64, 2, 32, 32, 256, 312, 1, stream);
    launch_conv3(Bb, wp[k][9], P[k][DU1B2], Ab, B, 32, 1, 32, 32, 256, 312, 1, stream);

    // 1x1 conv (no activation), Cout=28 in Cop=32 -> Bb
    launch_conv1(Ab, wp[k][10], P[k][LB], Bb, B, 32, 1, 28, 32, 256, 312, 0, stream);

    // x += shift(+1)(tanh(last))   (residual stays f32-exact)
    final_add_shift_kernel<<<nblk(n_x), EW_BLOCK, 0, stream>>>(Bb, xb, C, 32, H, W, WP, n_x);
  }

  hipMemcpyAsync(d_out, xb, (size_t)out_size * sizeof(float),
                 hipMemcpyDeviceToDevice, stream);
}
